// SparseMHA_21818433863964
// MI455X (gfx1250) — matrix operand order repaired and checked by reading, not compiled, not run on hardware
//
#include <hip/hip_runtime.h>

#define HIDDEN 512
#define NH 8

typedef __attribute__((ext_vector_type(16))) _Float16 v16h;
typedef __attribute__((ext_vector_type(8)))  _Float16 v8h;
typedef __attribute__((ext_vector_type(8)))  float    v8f;

// ---------- helpers: order-preserving float<->uint encoding for atomic max ----
__device__ __forceinline__ unsigned fenc(float f) {
    unsigned u = __float_as_uint(f);
    return (u & 0x80000000u) ? ~u : (u | 0x80000000u);
}
__device__ __forceinline__ float fdec(unsigned u) {
    u = (u & 0x80000000u) ? (u & 0x7fffffffu) : ~u;
    return __uint_as_float(u);
}

// ---------- utility kernels ----------
__global__ void zero_u32(unsigned* __restrict__ p, long n) {
    long i = (long)blockIdx.x * blockDim.x + threadIdx.x;
    if (i < n) p[i] = 0u;
}

__global__ void f32_to_f16(const float* __restrict__ src, _Float16* __restrict__ dst, long n) {
    long i = (long)blockIdx.x * blockDim.x + threadIdx.x;
    if (i < n) dst[i] = (_Float16)src[i];
}

// W (512x512 f32, row-major K x N) -> Wt (512x512 f16, row-major N x K)
__global__ void transpose512_f16(const float* __restrict__ W, _Float16* __restrict__ Wt) {
    int idx = blockIdx.x * blockDim.x + threadIdx.x;   // 512*512 threads
    int n = idx >> 9;
    int k = idx & 511;
    Wt[(size_t)n * HIDDEN + k] = (_Float16)W[(size_t)k * HIDDEN + n];
}

// ---------- fragment loaders ----------
__device__ __forceinline__ v16h load_a_frag(const _Float16* ap) {
    v8h alo = *(const v8h*)(ap);
    v8h ahi = *(const v8h*)(ap + 8);
    return __builtin_shufflevector(alo, ahi,
               0,1,2,3,4,5,6,7,8,9,10,11,12,13,14,15);
}

// ---------- WMMA GEMM: C[M x 512] = A[M x 512] * B[512 x 512] + bias ----------
// A: f16 row-major (M x K).  Bt: f16 row-major (N x K) == B transposed.
// One wave computes a 16x64 strip of C (4 tiles). Software-pipelined: the
// next K-step's fragments are loaded before the current step's 4 WMMAs, so
// loads stay outstanding while v_wmma_f32_16x16x32_f16 executes.
__global__ void __launch_bounds__(128) gemm_f16_wmma(
    const _Float16* __restrict__ A, const _Float16* __restrict__ Bt,
    const float* __restrict__ bias, float* __restrict__ C, int M)
{
    const int lane = threadIdx.x & 31;
    const int wv   = threadIdx.x >> 5;                 // 4 waves / block
    const int row0 = blockIdx.x * 16;
    const int col0 = (blockIdx.y * 4 + wv) * 64;       // 4 col tiles per wave
    const int mrow = lane & 15;
    const int hsel = lane >> 4;                        // half-wave select

    int arow = row0 + mrow;
    if (arow >= M) arow = M - 1;                       // clamp (stores guarded below)

    // A fragment K pattern (16-bit A 16x32): lanes 0-15 -> K 0..15,
    // lanes 16-31 -> K 16..31.
    const _Float16* ap = A  + (size_t)arow * HIDDEN + hsel * 16;
    // B fragment K pattern (16-bit B 32x16): lanes 0-15 -> K 0..15,
    // lanes 16-31 -> K 16..31, column (lane&15) per lane.
    const _Float16* bp = Bt + (size_t)(col0 + mrow) * HIDDEN + hsel * 16;

    v8f acc0 = {0.f,0.f,0.f,0.f,0.f,0.f,0.f,0.f};
    v8f acc1 = acc0, acc2 = acc0, acc3 = acc0;

    // ---- prologue: fragments for k0 = 0 ----
    v16h a  = load_a_frag(ap);
    v16h b0 = *(const v16h*)(bp);
    v16h b1 = *(const v16h*)(bp + 16 * HIDDEN);
    v16h b2 = *(const v16h*)(bp + 32 * HIDDEN);
    v16h b3 = *(const v16h*)(bp + 48 * HIDDEN);

    #pragma unroll 2
    for (int k0 = 32; k0 < HIDDEN; k0 += 32) {
        // issue next step's loads first (independent of current WMMAs)
        v16h an  = load_a_frag(ap + k0);
        v16h bn0 = *(const v16h*)(bp + k0);
        v16h bn1 = *(const v16h*)(bp + k0 + 16 * HIDDEN);
        v16h bn2 = *(const v16h*)(bp + k0 + 32 * HIDDEN);
        v16h bn3 = *(const v16h*)(bp + k0 + 48 * HIDDEN);

        acc0 = __builtin_amdgcn_wmma_f32_16x16x32_f16(false, a, false, b0, (short)0, acc0, false, false);
        acc1 = __builtin_amdgcn_wmma_f32_16x16x32_f16(false, a, false, b1, (short)0, acc1, false, false);
        acc2 = __builtin_amdgcn_wmma_f32_16x16x32_f16(false, a, false, b2, (short)0, acc2, false, false);
        acc3 = __builtin_amdgcn_wmma_f32_16x16x32_f16(false, a, false, b3, (short)0, acc3, false, false);

        a = an; b0 = bn0; b1 = bn1; b2 = bn2; b3 = bn3;
    }

    // ---- epilogue: last K-step ----
    acc0 = __builtin_amdgcn_wmma_f32_16x16x32_f16(false, a, false, b0, (short)0, acc0, false, false);
    acc1 = __builtin_amdgcn_wmma_f32_16x16x32_f16(false, a, false, b1, (short)0, acc1, false, false);
    acc2 = __builtin_amdgcn_wmma_f32_16x16x32_f16(false, a, false, b2, (short)0, acc2, false, false);
    acc3 = __builtin_amdgcn_wmma_f32_16x16x32_f16(false, a, false, b3, (short)0, acc3, false, false);

    // C layout: lanes 0-15 rows row0+0..7, lanes 16-31 rows row0+8..15, N = lane&15
    const int crow = row0 + hsel * 8;
    const int ccol = col0 + (lane & 15);

#define STORE_TILE(T, ACC)                                                     \
    {                                                                          \
        const int cc = ccol + 16 * (T);                                        \
        const float bb = bias[cc];                                             \
        _Pragma("unroll")                                                      \
        for (int j = 0; j < 8; ++j) {                                          \
            int r = crow + j;                                                  \
            if (r < M) C[(size_t)r * HIDDEN + cc] = (ACC)[j] + bb;             \
        }                                                                      \
    }
    STORE_TILE(0, acc0)
    STORE_TILE(1, acc1)
    STORE_TILE(2, acc2)
    STORE_TILE(3, acc3)
#undef STORE_TILE
}

// ---------- SDDMM + segment max: one wave per edge ----------
// head(col) == col % 8 == lane % 8, so cross-lane reduction over lanes with the
// same (lane&7) via shfl_xor(8) + shfl_xor(16) yields per-head scores.
__global__ void __launch_bounds__(256) sddmm_max(
    const float* __restrict__ q, const float* __restrict__ k,
    const int* __restrict__ rows, const int* __restrict__ cols,
    float* __restrict__ scores, unsigned* __restrict__ m_enc, int E)
{
    int e = (int)(((long)blockIdx.x * blockDim.x + threadIdx.x) >> 5);
    int lane = threadIdx.x & 31;
    if (e >= E) return;
    int r = rows[e], c = cols[e];
    const float* qr = q + (size_t)r * HIDDEN;
    const float* kc = k + (size_t)c * HIDDEN;
    __builtin_prefetch(qr, 0, 3);          // global_prefetch_b8: pull rows toward L1
    __builtin_prefetch(kc, 0, 3);
    float s = 0.f;
    #pragma unroll
    for (int i = 0; i < 16; ++i) {
        int col = lane + 32 * i;
        s += qr[col] * kc[col];
    }
    s += __shfl_xor(s, 8, 32);
    s += __shfl_xor(s, 16, 32);
    if (lane < 8) {
        scores[(size_t)e * NH + lane] = s;
        atomicMax(m_enc + (size_t)r * NH + lane, fenc(s));
    }
}

// ---------- exp + segment sum: one thread per (edge, head) ----------
__global__ void exp_sum(const int* __restrict__ rows, float* __restrict__ scores,
                        const unsigned* __restrict__ m_enc, float* __restrict__ denom,
                        long EN)
{
    long i = (long)blockIdx.x * blockDim.x + threadIdx.x;
    if (i >= EN) return;
    int e = (int)(i >> 3);
    int h = (int)(i & 7);
    int r = rows[e];
    float m  = fdec(m_enc[(size_t)r * NH + h]);
    float ex = __expf(scores[i] - m);
    scores[i] = ex;
    atomicAdd(denom + (size_t)r * NH + h, ex);
}

// ---------- SpMM scatter: one wave per edge ----------
__global__ void __launch_bounds__(256) spmm_scatter(
    const float* __restrict__ k, const int* __restrict__ rows,
    const int* __restrict__ cols, const float* __restrict__ ex,
    const float* __restrict__ denom, float* __restrict__ acc, int E)
{
    int e = (int)(((long)blockIdx.x * blockDim.x + threadIdx.x) >> 5);
    int lane = threadIdx.x & 31;
    if (e >= E) return;
    int r = rows[e], c = cols[e];
    const float* kc = k + (size_t)c * HIDDEN;
    __builtin_prefetch(kc, 0, 3);
    int h = lane & 7;
    float attn = ex[(size_t)e * NH + h] / denom[(size_t)r * NH + h];
    float* ar = acc + (size_t)r * HIDDEN;
    #pragma unroll
    for (int i = 0; i < 16; ++i) {
        int col = lane + 32 * i;
        atomicAdd(ar + col, attn * kc[col]);
    }
}

// ---------- host launch ----------
extern "C" void kernel_launch(void* const* d_in, const int* in_sizes, int n_in,
                              void* d_out, int out_size, void* d_ws, size_t ws_size,
                              hipStream_t stream) {
    const float* h  = (const float*)d_in[0];
    const float* Wq = (const float*)d_in[1];
    const float* bq = (const float*)d_in[2];
    const float* Wk = (const float*)d_in[3];
    const float* bk = (const float*)d_in[4];
    const float* Wo = (const float*)d_in[5];
    const float* bo = (const float*)d_in[6];
    const int* rows = (const int*)d_in[7];
    const int* cols = (const int*)d_in[8];
    float* out = (float*)d_out;

    const int N = in_sizes[0] / HIDDEN;   // 25000
    const int E = in_sizes[7];            // 400000

    // ---- workspace layout (all offsets 256B-aligned) ----
    char* w = (char*)d_ws;
    size_t off = 0;
    _Float16* h16 = (_Float16*)(w + off); off += (size_t)N * HIDDEN * 2;  // 25.6MB, reused for acc_f16
    off = (off + 255) & ~(size_t)255;
    _Float16* wqT = (_Float16*)(w + off); off += (size_t)HIDDEN * HIDDEN * 2;
    _Float16* wkT = (_Float16*)(w + off); off += (size_t)HIDDEN * HIDDEN * 2;
    _Float16* woT = (_Float16*)(w + off); off += (size_t)HIDDEN * HIDDEN * 2;
    off = (off + 255) & ~(size_t)255;
    float* q    = (float*)(w + off); off += (size_t)N * HIDDEN * 4;       // reused as acc
    float* kbuf = (float*)(w + off); off += (size_t)N * HIDDEN * 4;
    float* scr  = (float*)(w + off); off += (size_t)E * NH * 4;           // scores -> ex (in place)
    unsigned* m_enc = (unsigned*)(w + off); off += (size_t)N * NH * 4;
    float* denom    = (float*)(w + off); off += (size_t)N * NH * 4;       // contiguous after m_enc

    const long nh = (long)N * HIDDEN;

    // 1) conversions
    f32_to_f16<<<(unsigned)((nh + 255) / 256), 256, 0, stream>>>(h, h16, nh);
    transpose512_f16<<<(HIDDEN * HIDDEN) / 256, 256, 0, stream>>>(Wq, wqT);
    transpose512_f16<<<(HIDDEN * HIDDEN) / 256, 256, 0, stream>>>(Wk, wkT);
    transpose512_f16<<<(HIDDEN * HIDDEN) / 256, 256, 0, stream>>>(Wo, woT);

    // 2) q/k projections (WMMA): 4 waves/block, each wave a 16x64 strip
    dim3 ggrid((N + 15) / 16, HIDDEN / 256);
    gemm_f16_wmma<<<ggrid, 128, 0, stream>>>(h16, wqT, bq, q,    N);
    gemm_f16_wmma<<<ggrid, 128, 0, stream>>>(h16, wkT, bk, kbuf, N);

    // 3) zero segment state (m_enc + denom contiguous), every call
    long zn = (long)2 * N * NH;
    zero_u32<<<(unsigned)((zn + 255) / 256), 256, 0, stream>>>(m_enc, zn);

    // 4) SDDMM + segment max
    long et = (long)E * 32;
    sddmm_max<<<(unsigned)((et + 255) / 256), 256, 0, stream>>>(q, kbuf, rows, cols, scr, m_enc, E);

    // 5) exp + segment sum
    long en = (long)E * NH;
    exp_sum<<<(unsigned)((en + 255) / 256), 256, 0, stream>>>(rows, scr, m_enc, denom, en);

    // 6) zero accumulator (aliases q; q dead after SDDMM), then SpMM scatter
    zero_u32<<<(unsigned)((nh + 255) / 256), 256, 0, stream>>>((unsigned*)q, nh);
    spmm_scatter<<<(unsigned)((et + 255) / 256), 256, 0, stream>>>(kbuf, rows, cols, scr, denom, q, E);

    // 7) output projection (WMMA): acc -> f16 (reuse h16), then GEMM + bo -> out
    f32_to_f16<<<(unsigned)((nh + 255) / 256), 256, 0, stream>>>(q, h16, nh);
    gemm_f16_wmma<<<ggrid, 128, 0, stream>>>(h16, woT, bo, out, N);
}
